// SwinTransformerBlockV2_62680752718181
// MI455X (gfx1250) — compile-verified, matrix operand order checked by reading, not run on hardware
//
#include <hip/hip_runtime.h>
#include <hip/hip_bf16.h>
#include <math.h>

typedef __attribute__((ext_vector_type(16))) __bf16 v16bf;
typedef __attribute__((ext_vector_type(8)))  float  v8f;
typedef __attribute__((ext_vector_type(4)))  int    v4i;

#define M_TOK 50176

// ---- gfx1250 async global->LDS path (guarded so either toolchain compiles) ----
#if defined(__has_builtin)
#if __has_builtin(__builtin_amdgcn_global_load_async_to_lds_b128) && \
    __has_builtin(__builtin_amdgcn_s_wait_asynccnt)
#define USE_ASYNC_LDS 1
#endif
#endif
#ifndef USE_ASYNC_LDS
#define USE_ASYNC_LDS 0
#endif

__device__ __forceinline__ void async_wait0() {
#if USE_ASYNC_LDS
    __builtin_amdgcn_s_wait_asynccnt(0);
#endif
}

// copy 16 bf16 (32 bytes) global -> LDS; both pointers 16B-aligned
__device__ __forceinline__ void cp32_g2l(__bf16* lds, const __bf16* gl) {
#if USE_ASYNC_LDS
    typedef __attribute__((address_space(1))) v4i gv4i;
    typedef __attribute__((address_space(3))) v4i lv4i;
    __builtin_amdgcn_global_load_async_to_lds_b128((gv4i*)gl, (lv4i*)lds, 0, 0);
    __builtin_amdgcn_global_load_async_to_lds_b128((gv4i*)(gl + 8), (lv4i*)(lds + 8), 0, 0);
#else
    const uint4* s = (const uint4*)gl;
    uint4* d = (uint4*)lds;
    d[0] = s[0];
    d[1] = s[1];
#endif
}

// ---------- WMMA fragment helpers (CDNA5 16x16x32 bf16 layouts) ----------
__device__ __forceinline__ int kpair_of(int v, int g) {
    return (v < 4) ? (g * 8 + 2 * v) : (16 + g * 8 + 2 * (v - 4));
}

// A-style fragment (16 rows x 32 K bf16) from LDS; B operands are stored
// transposed (Bt[n][k]) so they use the identical lane layout.
__device__ __forceinline__ v16bf load_frag(const __bf16* base, int stride, int lane) {
    union { v16bf v; unsigned u[8]; } f;
    const int g = lane >> 4;
    const int m = lane & 15;
    const __bf16* rp = base + m * stride;
#pragma unroll
    for (int vv = 0; vv < 8; ++vv)
        f.u[vv] = *(const unsigned*)(rp + kpair_of(vv, g));
    return f.v;
}

// ---------- one-time converters ----------
__global__ __launch_bounds__(256)
void f32_to_bf16_kernel(const float* __restrict__ src, __bf16* __restrict__ dst, int n4) {
    const int idx = blockIdx.x * 256 + threadIdx.x;
    if (idx >= n4) return;
    float4 f = ((const float4*)src)[idx];
    dst[idx * 4 + 0] = (__bf16)f.x;
    dst[idx * 4 + 1] = (__bf16)f.y;
    dst[idx * 4 + 2] = (__bf16)f.z;
    dst[idx * 4 + 3] = (__bf16)f.w;
}

// Wt[n][k] = bf16(W[k][n])  (transpose + downconvert, one-time)
__global__ __launch_bounds__(256)
void wt_transpose_kernel(const float* __restrict__ W, __bf16* __restrict__ Wt,
                         int K, int N) {
    const int idx = blockIdx.x * 256 + threadIdx.x;
    if (idx >= K * N) return;
    const int k = idx / N, n = idx - k * N;
    Wt[(size_t)n * K + k] = (__bf16)W[idx];
}

// ---------- generic bf16 WMMA GEMM: C = act(A @ B + bias) ----------
// A: bf16 [M][K] (AMODE=1: rows gathered via swin shift/window map, K==512)
// Bt: bf16 [N][K] (pre-transposed weights)
// OMODE: 0 f32 out, 1 bf16 out, 2 exact-GELU + bf16 out
// BMODE: 0 bias0[col]; 1 concat(q_bias, 0, v_bias) over N=1536
template <int AMODE, int OMODE, int BMODE>
__global__ __launch_bounds__(256)
void gemm_bf16_wmma(const __bf16* __restrict__ Ab,
                    const __bf16* __restrict__ Bt,
                    const float* __restrict__ bias0,
                    const float* __restrict__ bias1,
                    void* __restrict__ Cptr,
                    int M, int N, int K) {
    __shared__ __align__(16) __bf16 As[2][128][40];
    __shared__ __align__(16) __bf16 Bs[2][128][40];

    const int tid  = threadIdx.x;
    const int lane = tid & 31;
    const int wv   = tid >> 5;
    const int wm   = wv & 3;   // M sub-tile (32 rows)
    const int wn   = wv >> 2;  // N sub-tile (64 cols)
    const int blockRow = blockIdx.x * 128;
    const int blockCol = blockIdx.y * 128;

    // cooperative-load assignment: 2 threads per row/col, 16 bf16 each
    const int ar  = tid >> 1;
    const int akc = (tid & 1) * 16;
    const __bf16* asrc;
    if (AMODE == 1) {
        const int r  = blockRow + ar;
        const int w  = r / 49, p = r % 49;
        const int b  = w >> 6, wl = w & 63;
        const int wh = wl >> 3, wwi = wl & 7;
        const int i  = p / 7,  j  = p % 7;
        const int sh = (wh * 7 + i + 3) % 56;   // roll(-SHIFT)
        const int sw = (wwi * 7 + j + 3) % 56;
        asrc = Ab + (size_t)((b * 56 + sh) * 56 + sw) * 512 + akc;
    } else {
        asrc = Ab + (size_t)(blockRow + ar) * K + akc;
    }
    const __bf16* bsrc = Bt + (size_t)(blockCol + (tid >> 1)) * K + akc;

    v8f c[2][4];
#pragma unroll
    for (int i = 0; i < 2; ++i)
#pragma unroll
        for (int j = 0; j < 4; ++j) c[i][j] = (v8f)0.0f;

    // prologue: first K-tile into buffer 0
    cp32_g2l(&As[0][ar][akc], asrc);
    cp32_g2l(&Bs[0][ar][akc], bsrc);

    int cur = 0;
    for (int k0 = 0; k0 < K; k0 += 32) {
        async_wait0();
        __syncthreads();
        if (k0 + 32 < K) {  // prefetch next K-tile into other buffer
            cp32_g2l(&As[cur ^ 1][ar][akc], asrc + k0 + 32);
            cp32_g2l(&Bs[cur ^ 1][ar][akc], bsrc + k0 + 32);
            __builtin_prefetch(asrc + k0 + 64, 0, 1);
            __builtin_prefetch(bsrc + k0 + 64, 0, 1);
        }

        v16bf a[2], b[4];
#pragma unroll
        for (int mt = 0; mt < 2; ++mt)
            a[mt] = load_frag(&As[cur][wm * 32 + mt * 16][0], 40, lane);
#pragma unroll
        for (int nt = 0; nt < 4; ++nt)
            b[nt] = load_frag(&Bs[cur][wn * 64 + nt * 16][0], 40, lane);
#pragma unroll
        for (int mt = 0; mt < 2; ++mt)
#pragma unroll
            for (int nt = 0; nt < 4; ++nt)
                c[mt][nt] = __builtin_amdgcn_wmma_f32_16x16x32_bf16(
                    false, a[mt], false, b[nt], (short)0, c[mt][nt], false, false);
        cur ^= 1;
    }

    const int g = lane >> 4, nl = lane & 15;
#pragma unroll
    for (int mt = 0; mt < 2; ++mt)
#pragma unroll
        for (int nt = 0; nt < 4; ++nt)
#pragma unroll
            for (int r = 0; r < 8; ++r) {
                const int row = blockRow + wm * 32 + mt * 16 + r + 8 * g;
                const int col = blockCol + wn * 64 + nt * 16 + nl;
                float bv;
                if (BMODE == 1)
                    bv = (col < 512) ? bias0[col]
                                     : (col < 1024 ? 0.f : bias1[col - 1024]);
                else
                    bv = bias0[col];
                float v = c[mt][nt][r] + bv;
                if (OMODE == 2) v = 0.5f * v * (1.f + erff(v * 0.70710678118f));
                if (OMODE == 0)
                    ((float*)Cptr)[(size_t)row * N + col] = v;
                else
                    ((__bf16*)Cptr)[(size_t)row * N + col] = (__bf16)v;
            }
}

// ---------- CPB MLP: table[169][16] = relu(t @ w1 + b1) @ w2 ----------
__global__ __launch_bounds__(256)
void cpb_mlp_kernel(const float* __restrict__ tbl, const float* __restrict__ w1,
                    const float* __restrict__ b1,  const float* __restrict__ w2,
                    float* __restrict__ out) {
    __shared__ float hid[512];
    const int e = blockIdx.x;
    const float t0 = tbl[e * 2], t1 = tbl[e * 2 + 1];
    for (int j = threadIdx.x; j < 512; j += 256)
        hid[j] = fmaxf(t0 * w1[j] + t1 * w1[512 + j] + b1[j], 0.f);
    __syncthreads();
    if (threadIdx.x < 16) {
        const int h = threadIdx.x;
        float s = 0.f;
        for (int j = 0; j < 512; ++j) s += hid[j] * w2[j * 16 + h];
        out[e * 16 + h] = s;
    }
}

// biasArr[h][i][j] = 16*sigmoid(table[rpi[i*49+j]][h])
__global__ void bias_gather_kernel(const float* __restrict__ table,
                                   const int* __restrict__ rpi,
                                   float* __restrict__ biasArr) {
    const int idx = blockIdx.x * 256 + threadIdx.x;
    if (idx >= 16 * 2401) return;
    const int h = idx / 2401, ij = idx % 2401;
    const float v = table[rpi[ij] * 16 + h];
    biasArr[idx] = 16.f / (1.f + __expf(-v));
}

// ---------- attention: one block per (window, head) ----------
__global__ __launch_bounds__(128)
void attn_kernel(const __bf16* __restrict__ qkv,        // [M_TOK][1536]
                 const float* __restrict__ logit_scale, // [16]
                 const float* __restrict__ biasArr,     // [16][49][49]
                 const float* __restrict__ attn_mask,   // [64][49][49]
                 __bf16* __restrict__ attn_out) {       // [M_TOK][512]
    __shared__ __align__(16) __bf16 Qs[64][34];
    __shared__ __align__(16) __bf16 Ks[64][34];
    __shared__ __align__(16) __bf16 Vt[32][66];   // Vt[d][k]
    __shared__ float  Sf[64][64];
    __shared__ __align__(16) __bf16 P[64][66];

    const int bid = blockIdx.x;
    const int w = bid >> 4, h = bid & 15;
    const int tid = threadIdx.x, lane = tid & 31, wv = tid >> 5;
    const size_t qb = (size_t)w * 49 * 1536;
    const int wi = w & 63;

    if (tid < 98) {  // L2-normalize q (t<49) / k rows
        const int isk = tid >= 49;
        const int i = isk ? tid - 49 : tid;
        const __bf16* src = qkv + qb + (size_t)i * 1536 + (isk ? 512 : 0) + h * 32;
        float tmp[32], ss = 0.f;
#pragma unroll
        for (int d = 0; d < 32; ++d) { float f = (float)src[d]; tmp[d] = f; ss += f * f; }
        const float inv = rsqrtf(ss + 1e-20f);
        __bf16* dst = isk ? &Ks[i][0] : &Qs[i][0];
#pragma unroll
        for (int d = 0; d < 32; ++d) dst[d] = (__bf16)(tmp[d] * inv);
    }
    for (int idx = tid; idx < 15 * 32; idx += 128) {  // zero pad rows 49..63
        const int i = 49 + idx / 32, d = idx % 32;
        Qs[i][d] = (__bf16)0.f; Ks[i][d] = (__bf16)0.f;
    }
    for (int idx = tid; idx < 64 * 32; idx += 128) {  // V transposed, pad=0
        const int kk = idx >> 5, d = idx & 31;
        Vt[d][kk] = (kk < 49) ? qkv[qb + (size_t)kk * 1536 + 1024 + h * 32 + d]
                              : (__bf16)0.f;
    }
    // stage bias+mask into Sf (padded region = -1e30); makes the WMMA
    // epilogue an unconditional LDS RMW (padded cc is exactly 0)
    for (int idx = tid; idx < 64 * 64; idx += 128) {
        const int i = idx >> 6, j = idx & 63;
        float v = -1e30f;
        if (i < 49 && j < 49)
            v = biasArr[(h * 49 + i) * 49 + j] +
                attn_mask[((size_t)wi * 49 + i) * 49 + j];
        Sf[i][j] = v;
    }
    __syncthreads();

    const float scale = __expf(fminf(logit_scale[h], 4.60517019f));  // ln(100)
    {
        const v16bf a = load_frag(&Qs[wv * 16][0], 34, lane);
        const int g = lane >> 4, nl = lane & 15;
#pragma unroll
        for (int nt = 0; nt < 4; ++nt) {
            const v16bf b = load_frag(&Ks[nt * 16][0], 34, lane);
            v8f cc = (v8f)0.0f;
            cc = __builtin_amdgcn_wmma_f32_16x16x32_bf16(
                false, a, false, b, (short)0, cc, false, false);
#pragma unroll
            for (int r = 0; r < 8; ++r) {
                const int i = wv * 16 + r + 8 * g;
                const int j = nt * 16 + nl;
                Sf[i][j] = cc[r] * scale + Sf[i][j];
            }
        }
    }
    __syncthreads();

    if (tid < 64) {  // softmax rows (multi-pass, no spill)
        const int i = tid;
        if (i < 49) {
            float mx = -1e30f;
            for (int j = 0; j < 49; ++j) mx = fmaxf(mx, Sf[i][j]);
            float s = 0.f;
            for (int j = 0; j < 49; ++j) s += __expf(Sf[i][j] - mx);
            const float inv = 1.f / s;
            for (int j = 0; j < 49; ++j) P[i][j] = (__bf16)(__expf(Sf[i][j] - mx) * inv);
            for (int j = 49; j < 64; ++j) P[i][j] = (__bf16)0.f;
        } else {
            for (int j = 0; j < 64; ++j) P[i][j] = (__bf16)0.f;
        }
    }
    __syncthreads();

    {   // O = P @ V : M=64(pad), N=32, K=64(pad)
        v8f o[2]; o[0] = (v8f)0.0f; o[1] = (v8f)0.0f;
#pragma unroll
        for (int kc = 0; kc < 2; ++kc) {
            const v16bf a = load_frag(&P[wv * 16][kc * 32], 66, lane);
#pragma unroll
            for (int nt = 0; nt < 2; ++nt) {
                const v16bf b = load_frag(&Vt[nt * 16][kc * 32], 66, lane);
                o[nt] = __builtin_amdgcn_wmma_f32_16x16x32_bf16(
                    false, a, false, b, (short)0, o[nt], false, false);
            }
        }
        const int g = lane >> 4, nl = lane & 15;
#pragma unroll
        for (int nt = 0; nt < 2; ++nt)
#pragma unroll
            for (int r = 0; r < 8; ++r) {
                const int i = wv * 16 + r + 8 * g;
                const int d = nt * 16 + nl;
                if (i < 49)
                    attn_out[(size_t)(w * 49 + i) * 512 + h * 32 + d] = (__bf16)o[nt][r];
            }
    }
}

// ---------- residual + LayerNorm (wave per token) ----------
template <int MAPPED, int WRITE_BF16>
__global__ __launch_bounds__(256)
void ln_residual_kernel(const float* __restrict__ resid,
                        const float* __restrict__ y,
                        const float* __restrict__ gamma,
                        const float* __restrict__ beta,
                        float* __restrict__ outf,
                        __bf16* __restrict__ outb) {
    const int t = blockIdx.x * 8 + (threadIdx.x >> 5);
    const int lane = threadIdx.x & 31;
    int row;
    if (MAPPED) {
        const int b = t / 3136, rem = t - b * 3136;
        const int hh = rem / 56, ww = rem % 56;
        const int sh = (hh + 53) % 56, sw = (ww + 53) % 56;  // roll(+3) inverse
        row = (b * 64 + (sh / 7) * 8 + (sw / 7)) * 49 + (sh % 7) * 7 + (sw % 7);
    } else {
        row = t;
    }
    const float* yp = y + (size_t)row * 512 + lane * 16;
    float v[16], s = 0.f, ss = 0.f;
#pragma unroll
    for (int q = 0; q < 4; ++q) {
        float4 f = ((const float4*)yp)[q];
        v[q * 4 + 0] = f.x; v[q * 4 + 1] = f.y;
        v[q * 4 + 2] = f.z; v[q * 4 + 3] = f.w;
    }
#pragma unroll
    for (int i = 0; i < 16; ++i) { s += v[i]; ss += v[i] * v[i]; }
#pragma unroll
    for (int m = 1; m < 32; m <<= 1) {
        s  += __shfl_xor(s, m, 32);
        ss += __shfl_xor(ss, m, 32);
    }
    const float mu  = s * (1.f / 512.f);
    const float var = ss * (1.f / 512.f) - mu * mu;
    const float inv = rsqrtf(var + 1e-5f);
    const size_t ob = (size_t)t * 512 + lane * 16;
#pragma unroll
    for (int i = 0; i < 16; ++i) {
        const int ch = lane * 16 + i;
        const float ln = (v[i] - mu) * inv * gamma[ch] + beta[ch];
        const float o  = resid[ob + i] + ln;
        outf[ob + i] = o;
        if (WRITE_BF16) outb[ob + i] = (__bf16)o;
    }
}

extern "C" void kernel_launch(void* const* d_in, const int* in_sizes, int n_in,
                              void* d_out, int out_size, void* d_ws, size_t ws_size,
                              hipStream_t stream) {
    const float* x         = (const float*)d_in[0];
    const float* qkv_w     = (const float*)d_in[1];
    const float* q_bias    = (const float*)d_in[2];
    const float* v_bias    = (const float*)d_in[3];
    const float* proj_w    = (const float*)d_in[4];
    const float* proj_b    = (const float*)d_in[5];
    const float* logit_sc  = (const float*)d_in[6];
    const float* cpb_w1    = (const float*)d_in[7];
    const float* cpb_b1    = (const float*)d_in[8];
    const float* cpb_w2    = (const float*)d_in[9];
    const float* rct       = (const float*)d_in[10];
    const float* gamma1    = (const float*)d_in[11];
    const float* beta1     = (const float*)d_in[12];
    const float* gamma2    = (const float*)d_in[13];
    const float* beta2     = (const float*)d_in[14];
    const float* fc1_w     = (const float*)d_in[15];
    const float* fc1_b     = (const float*)d_in[16];
    const float* fc2_w     = (const float*)d_in[17];
    const float* fc2_b     = (const float*)d_in[18];
    const int*   rpi       = (const int*)d_in[19];
    const float* attn_mask = (const float*)d_in[20];

    char* ws = (char*)d_ws;
    // region0 hosts qkv (bf16, 154MB) then hidden (bf16, 205MB): disjoint lifetimes
    __bf16* qkv    = (__bf16*)(ws + 0);
    __bf16* hidden = (__bf16*)(ws + 0);
    float*  ybuf   = (float*)(ws + 205520896);   // proj out, later fc2 out
    __bf16* attno  = (__bf16*)(ws + 308281344);
    float*  x1f    = (float*)(ws + 359661568);
    __bf16* x1b    = (__bf16*)(ws + 462422016);
    float*  table  = (float*)(ws + 513802240);
    float*  biasA  = (float*)(ws + 513813056);
    __bf16* xb     = (__bf16*)(ws + 513966720);  // bf16 copy of x
    __bf16* wt_qkv = (__bf16*)(ws + 565346944);  // [1536][512]
    __bf16* wt_prj = (__bf16*)(ws + 566919808);  // [512][512]
    __bf16* wt_fc1 = (__bf16*)(ws + 567444096);  // [2048][512]
    __bf16* wt_fc2 = (__bf16*)(ws + 569541248);  // [512][2048]

    // 0) one-time conversions: x -> bf16; weights -> bf16 transposed [N][K]
    f32_to_bf16_kernel<<<25088, 256, 0, stream>>>(x, xb, M_TOK * 512 / 4);
    wt_transpose_kernel<<<(512 * 1536 + 255) / 256, 256, 0, stream>>>(qkv_w, wt_qkv, 512, 1536);
    wt_transpose_kernel<<<(512 * 512 + 255) / 256, 256, 0, stream>>>(proj_w, wt_prj, 512, 512);
    wt_transpose_kernel<<<(512 * 2048 + 255) / 256, 256, 0, stream>>>(fc1_w, wt_fc1, 512, 2048);
    wt_transpose_kernel<<<(2048 * 512 + 255) / 256, 256, 0, stream>>>(fc2_w, wt_fc2, 2048, 512);
    // 1) continuous position bias
    cpb_mlp_kernel<<<169, 256, 0, stream>>>(rct, cpb_w1, cpb_b1, cpb_w2, table);
    bias_gather_kernel<<<(16 * 2401 + 255) / 256, 256, 0, stream>>>(table, rpi, biasA);
    // 2) shift + window-partition + QKV GEMM (+concat bias)
    gemm_bf16_wmma<1, 1, 1><<<dim3(392, 12), 256, 0, stream>>>(
        xb, wt_qkv, q_bias, v_bias, qkv, M_TOK, 1536, 512);
    // 3) cosine attention per (window, head)
    attn_kernel<<<1024 * 16, 128, 0, stream>>>(qkv, logit_sc, biasA, attn_mask, attno);
    // 4) output projection
    gemm_bf16_wmma<0, 0, 0><<<dim3(392, 4), 256, 0, stream>>>(
        attno, wt_prj, proj_b, nullptr, ybuf, M_TOK, 512, 512);
    // 5) window-reverse + unshift + residual + LN1
    ln_residual_kernel<1, 1><<<6272, 256, 0, stream>>>(
        x, ybuf, gamma1, beta1, x1f, x1b);
    // 6) fc1 + exact GELU
    gemm_bf16_wmma<0, 2, 0><<<dim3(392, 16), 256, 0, stream>>>(
        x1b, wt_fc1, fc1_b, nullptr, hidden, M_TOK, 2048, 512);
    // 7) fc2
    gemm_bf16_wmma<0, 0, 0><<<dim3(392, 4), 256, 0, stream>>>(
        hidden, wt_fc2, fc2_b, nullptr, ybuf, M_TOK, 512, 2048);
    // 8) residual + LN2 -> final output
    ln_residual_kernel<0, 0><<<6272, 256, 0, stream>>>(
        x1f, ybuf, gamma2, beta2, (float*)d_out, nullptr);
}
